// MazeBrain_61538291417133
// MI455X (gfx1250) — matrix operand order repaired and checked by reading, not compiled, not run on hardware
//
#include <hip/hip_runtime.h>

// SNN persistent kernel for MI455X (gfx1250, wave32, WMMA).
// T=1200 serial steps; per step: 16x1024x1024 GEMM (spikes @ W_rec) + LIF update.
// Strategy: 32 WGs x 64 threads (2 waves). Each wave owns a 16-neuron output tile.
// W_rec staged once into LDS (f16, WMMA-B swizzled, 64KB/WG). Spikes exchanged via
// a 32KB f16 double buffer in d_ws; one global spin barrier per step.

typedef __attribute__((ext_vector_type(16))) _Float16 v16h;
typedef __attribute__((ext_vector_type(8)))  _Float16 v8h;
typedef __attribute__((ext_vector_type(8)))  float    v8f;

#define T_STEPS 1200
#define BATCH   16
#define NN      1024
#define WHIST   1000
#define NWG     32
#define WAVES_PER_WG 2
#define THREADS (WAVES_PER_WG * 32)

// neuron constants
#define K_THRESHOLD (-55.0f)
#define K_REST      (-65.0f)
#define K_RESET     (-70.0f)
#define K_LEAK      (0.9f)
#define K_ADAPT_C   (0.2f)
#define K_ADAPT_DEC (0.95f)
#define K_REFRAC    (5.0f)
#define K_HOMEO_LR  (0.001f)
#define K_TARGET    (0.01f)

// d_ws layout: [0..3] barrier counter; spk f16 double buffer at +256 (2 x 32KB)
#define WS_SPK0 256
#define WS_SPK1 (256 + BATCH * NN * 2)

__global__ void snn_init(unsigned* cnt) { *cnt = 0u; }

__launch_bounds__(THREADS, 1)
__global__ void snn_persistent(const float* __restrict__ Iext,
                               const float* __restrict__ Wrec,
                               float* __restrict__ out,
                               unsigned char* __restrict__ wsb)
{
    // 2 waves * 32 k-blocks * 512 halves (1KB per (tile,kb) chunk) = 64KB
    __shared__ _Float16 Wlds[WAVES_PER_WG * 32 * 512];

    unsigned* barrier_cnt = (unsigned*)wsb;
    _Float16* spk0 = (_Float16*)(wsb + WS_SPK0);
    _Float16* spk1 = (_Float16*)(wsb + WS_SPK1);

    const int tidx  = threadIdx.x;
    const int wave  = tidx >> 5;
    const int lane  = tidx & 31;
    const int tile  = blockIdx.x * WAVES_PER_WG + wave;  // 0..63
    const int n0    = tile * 16;
    const int mcol  = lane & 15;       // A-row (batch) selector / B-column selector
    const int khalf = lane >> 4;       // 0 or 1 (K-half within fragment)
    const int ncol  = n0 + mcol;       // this lane's output-neuron column

    // ---- stage W_rec into LDS once, pre-swizzled into WMMA B-fragment layout ----
    // B 32x16 f16 frag: lanes 0-15 hold K=kb*32+{0..7,16..23} of column N=lane,
    //                   lanes 16-31 hold K=kb*32+{8..15,24..31} of column N=lane-16.
    _Float16* myW = &Wlds[wave * 32 * 512];
    for (int kb = 0; kb < 32; ++kb) {
        const int kbase = kb * 32 + khalf * 8;
        v8h lo, hi;
        #pragma unroll
        for (int j = 0; j < 8; ++j) lo[j] = (_Float16)Wrec[(kbase + j) * NN + ncol];
        #pragma unroll
        for (int j = 0; j < 8; ++j) hi[j] = (_Float16)Wrec[(kbase + 16 + j) * NN + ncol];
        const int off = kb * 512 + lane * 16;
        *(v8h*)&myW[off]     = lo;
        *(v8h*)&myW[off + 8] = hi;
    }
    __syncthreads();

    // ---- per-thread neuron state in WMMA C-fragment layout ----
    // C frag: VGPR r, lane l -> batch b = r + 8*(l>=16), neuron n = n0 + (l&15)
    float vmem[8], adapt[8], refr[8], thr[8], rsum[8];
    int   bn[8];
    #pragma unroll
    for (int r = 0; r < 8; ++r) {
        vmem[r]  = K_REST;
        adapt[r] = 0.0f;
        refr[r]  = 0.0f;
        thr[r]   = K_THRESHOLD;
        rsum[r]  = 0.0f;
        bn[r]    = (r + 8 * khalf) * NN + ncol;
    }

    float* spikes_out = out;
    float* pot_out    = out + (size_t)T_STEPS * BATCH * NN;

    for (int t = 0; t < T_STEPS; ++t) {
        // ---- recurrent input: acc = spikes_{t-1} @ W_rec (this tile's columns) ----
        v8f acc = {0.f, 0.f, 0.f, 0.f, 0.f, 0.f, 0.f, 0.f};
        if (t > 0) {
            const _Float16* spk = ((t - 1) & 1) ? spk1 : spk0;
            // A 16x32 f16 frag: lanes 0-15: M=lane, K = kb*32 + {0..7, 16..23}
            //                   lanes 16-31: M=lane-16, K = kb*32 + {8..15, 24..31}
            const _Float16* arow = spk + mcol * NN + khalf * 8;
            for (int kb = 0; kb < 32; ++kb) {
                union { v16h v; v8h h[2]; } A, Bm;
                A.h[0] = *(const v8h*)(arow + kb * 32);
                A.h[1] = *(const v8h*)(arow + kb * 32 + 16);
                const int off = kb * 512 + lane * 16;
                Bm.h[0] = *(const v8h*)&myW[off];
                Bm.h[1] = *(const v8h*)&myW[off + 8];
                acc = __builtin_amdgcn_wmma_f32_16x16x32_f16(
                          false, A.v, false, Bm.v, (short)0, acc, false, false);
            }
        }

        // ---- fused LIF + homeostasis update, all state in registers ----
        const float* Ie     = Iext       + (size_t)t * BATCH * NN;
        float*       so     = spikes_out + (size_t)t * BATCH * NN;
        float*       po     = pot_out    + (size_t)t * BATCH * NN;
        const float* so_old = spikes_out + (size_t)(t - WHIST) * BATCH * NN;
        _Float16*    sb     = (t & 1) ? spk1 : spk0;

        #pragma unroll
        for (int r = 0; r < 8; ++r) {
            const float I      = acc[r] + Ie[bn[r]];
            const bool  active = refr[r] <= 0.0f;
            const float vint   = K_LEAK * (vmem[r] - K_REST) + K_REST + I - adapt[r];
            const float v1     = active ? vint : vmem[r];
            const float refr1  = active ? refr[r] : refr[r] - 1.0f;
            const float spike  = (v1 >= thr[r]) ? 1.0f : 0.0f;
            const bool  fired  = spike > 0.0f;
            const float v2     = fired ? K_RESET : v1;
            adapt[r] = (fired ? adapt[r] + K_ADAPT_C : adapt[r]) * K_ADAPT_DEC;
            refr[r]  = fired ? K_REFRAC : refr1;
            const float old = (t >= WHIST) ? so_old[bn[r]] : 0.0f;  // circular buf == our own old output
            rsum[r] += spike - old;
            if (t + 1 >= WHIST) {
                float ta = thr[r] + K_HOMEO_LR * (rsum[r] * (1.0f / WHIST) - K_TARGET);
                thr[r] = fminf(fmaxf(ta, 0.1f), 5.0f);
            }
            vmem[r]    = v2;
            so[bn[r]]  = spike;
            po[bn[r]]  = v2;
            sb[bn[r]]  = (_Float16)spike;   // f16 spike exchange buffer (double-buffered)
        }

        // ---- device-wide barrier: one per step (double buffering makes one enough) ----
        __syncthreads();
        if (tidx == 0) {
            __threadfence();  // release our spike/state writes to device scope
            __hip_atomic_fetch_add(barrier_cnt, 1u, __ATOMIC_ACQ_REL, __HIP_MEMORY_SCOPE_AGENT);
            const unsigned target = (unsigned)NWG * (unsigned)(t + 1);
            while (__hip_atomic_load(barrier_cnt, __ATOMIC_ACQUIRE, __HIP_MEMORY_SCOPE_AGENT) < target)
                __builtin_amdgcn_s_sleep(1);
        }
        __syncthreads();
        __threadfence();      // acquire: see all other WGs' spike writes
    }
}

extern "C" void kernel_launch(void* const* d_in, const int* in_sizes, int n_in,
                              void* d_out, int out_size, void* d_ws, size_t ws_size,
                              hipStream_t stream) {
    const float* Iext = (const float*)d_in[0];   // [T, B, N] f32
    const float* Wrec = (const float*)d_in[1];   // [N, N]    f32
    float*       out  = (float*)d_out;           // spikes [T,B,N] then potentials [T,B,N]
    unsigned char* wsb = (unsigned char*)d_ws;

    snn_init<<<1, 1, 0, stream>>>((unsigned*)wsb);
    snn_persistent<<<NWG, THREADS, 0, stream>>>(Iext, Wrec, out, wsb);
}
